// MoELayer_10204842295374
// MI455X (gfx1250) — matrix operand order repaired and checked by reading, not compiled, not run on hardware
//
#include <hip/hip_runtime.h>

// ---------------------------------------------------------------------------
// MoE layer for MI455X (gfx1250), compile-only tuned.
//
// Shapes: T=8192 tokens, D=1024, H=4096, E=8, top-2 routing.
// Strategy (rooflined for MI455X):
//   * Routed FLOPs = 275 GFLOP (vs 1.1 TFLOP dense) -> gather-based routing.
//   * bf16 WMMA (v_wmma_f32_16x16x32_bf16), f32 accumulate.
//   * Weights pre-converted f32->bf16 AND transposed ([N][K] layout) into the
//     workspace: 134 MB total, fits the 192 MB L2, and makes every WMMA
//     B-fragment two contiguous 16B global_load_b128 per lane.
//   * Fused FFN block: 16-token tile x 1 expert; x-tile (32 KB bf16) and the
//     full h[16][4096] intermediate (128 KB bf16) live in LDS (160 KB of the
//     320 KB CDNA5 WGP LDS), so h never touches HBM.
//   * Epilogue: per-row renormalized top-2 weight * (y + b2), scatter-added
//     with global_atomic_add_f32 (exactly 2 commutative adds per element ->
//     deterministic).
// Workspace need: ~129 MiB (bf16 weights + routing tables).
// ---------------------------------------------------------------------------

#define TOK   8192
#define DDIM  1024
#define HDIM  4096
#define NEXP  8

typedef __attribute__((ext_vector_type(16))) __bf16       v16bf;
typedef __attribute__((ext_vector_type(8)))  float        v8f;
typedef __attribute__((ext_vector_type(4)))  unsigned int v4u;

union Frag {
    v16bf v;
    v4u   u[2];
};

__device__ inline v8f wmma_bf16(v16bf a, v16bf b, v8f c) {
    // 8 args: (neg_a, A, neg_b, B, c_mod, C, reuse_a, reuse_b)
    return __builtin_amdgcn_wmma_f32_16x16x32_bf16(
        false, a, false, b, (short)0, c, false, false);
}

// ---------------------------------------------------------------------------
// Zero the `out` region of d_out (required every call: epilogue accumulates).
// ---------------------------------------------------------------------------
__global__ __launch_bounds__(256) void zero_out_kernel(float* __restrict__ out) {
    const size_t i = (size_t)blockIdx.x * 256 + threadIdx.x;   // one float4 each
    float4 z; z.x = 0.f; z.y = 0.f; z.z = 0.f; z.w = 0.f;
    ((float4*)out)[i] = z;
}

// ---------------------------------------------------------------------------
// Tiled transpose + f32->bf16: in[K][N] (row-major) -> out[N][K] (row-major).
// blockIdx.z selects the expert slice. Block = (32, 8).
// ---------------------------------------------------------------------------
__global__ __launch_bounds__(256) void transpose_bf16_kernel(
    const float* __restrict__ in, __bf16* __restrict__ outp, int K, int N) {
    __shared__ __bf16 tile[32][33];
    const float*  src = in   + (size_t)blockIdx.z * K * N;
    __bf16*       dst = outp + (size_t)blockIdx.z * K * N;
    const int nt = blockIdx.x * 32;
    const int kt = blockIdx.y * 32;
    const int tx = threadIdx.x, ty = threadIdx.y;
#pragma unroll
    for (int i = 0; i < 4; ++i) {
        const int k = kt + ty + i * 8;
        tile[ty + i * 8][tx] = (__bf16)src[(size_t)k * N + nt + tx];  // coalesced read
    }
    __syncthreads();
#pragma unroll
    for (int i = 0; i < 4; ++i) {
        const int n = nt + ty + i * 8;
        dst[(size_t)n * K + kt + tx] = tile[tx][ty + i * 8];          // coalesced write
    }
}

// ---------------------------------------------------------------------------
// Gating: one wave32 per token. logits -> softmax -> top-2 -> softmax over the
// two probs (exactly what the reference does). Writes gate_probs and the
// per-token (idx0, idx1) / (w0, w1) routing tables.
// ---------------------------------------------------------------------------
__global__ __launch_bounds__(256) void gate_kernel(
    const float* __restrict__ x, const float* __restrict__ gW,
    const float* __restrict__ gb, float* __restrict__ gate_probs,
    int2* __restrict__ idx2, float2* __restrict__ w2v) {
    const int t    = blockIdx.x * 8 + (threadIdx.x >> 5);
    const int lane = threadIdx.x & 31;

    float acc[NEXP];
#pragma unroll
    for (int e = 0; e < NEXP; ++e) acc[e] = 0.f;

    const float* xr = x + (size_t)t * DDIM;
    for (int d = lane; d < DDIM; d += 32) {
        const float xv = xr[d];
        const float* g = gW + (size_t)d * NEXP;
#pragma unroll
        for (int e = 0; e < NEXP; ++e) acc[e] += xv * g[e];
    }
#pragma unroll
    for (int off = 16; off >= 1; off >>= 1) {
#pragma unroll
        for (int e = 0; e < NEXP; ++e) acc[e] += __shfl_xor(acc[e], off, 32);
    }

    if (lane == 0) {
        float l[NEXP], p[NEXP];
        float m = -3.4e38f;
#pragma unroll
        for (int e = 0; e < NEXP; ++e) { l[e] = acc[e] + gb[e]; m = fmaxf(m, l[e]); }
        float s = 0.f;
#pragma unroll
        for (int e = 0; e < NEXP; ++e) { p[e] = __expf(l[e] - m); s += p[e]; }
        const float inv = 1.f / s;
#pragma unroll
        for (int e = 0; e < NEXP; ++e) { p[e] *= inv; gate_probs[(size_t)t * NEXP + e] = p[e]; }
        // top-2 (ties -> lowest index, matching lax.top_k)
        int i0 = 0;
#pragma unroll
        for (int e = 1; e < NEXP; ++e) if (p[e] > p[i0]) i0 = e;
        int i1 = (i0 == 0) ? 1 : 0;
#pragma unroll
        for (int e = 0; e < NEXP; ++e) if (e != i0 && p[e] > p[i1]) i1 = e;
        // reference: softmax over the two top PROBS
        const float e0 = __expf(p[i0]), e1 = __expf(p[i1]);
        const float w0 = e0 / (e0 + e1);
        idx2[t] = make_int2(i0, i1);
        w2v[t]  = make_float2(w0, 1.f - w0);
    }
}

// ---------------------------------------------------------------------------
// Deterministic bucketing: one block per expert scans all tokens in order,
// block-wide prefix sum, appends (token, weight) to that expert's list.
// ---------------------------------------------------------------------------
__global__ __launch_bounds__(256) void bucket_kernel(
    const int2* __restrict__ idx2, const float2* __restrict__ w2v,
    int* __restrict__ counts, int* __restrict__ tlist, float* __restrict__ wlist) {
    const int e   = blockIdx.x;
    const int tid = threadIdx.x;
    __shared__ int scan[256];
    int base = 0;
    for (int c = 0; c < TOK; c += 256) {
        const int   t = c + tid;
        const int2  a = idx2[t];
        const float2 w = w2v[t];
        const int p0 = (a.x == e) ? 1 : 0;
        const int p1 = (a.y == e) ? 1 : 0;
        const int pred = p0 | p1;
        scan[tid] = pred;
        __syncthreads();
        for (int off = 1; off < 256; off <<= 1) {
            const int v = (tid >= off) ? scan[tid - off] : 0;
            __syncthreads();
            scan[tid] += v;
            __syncthreads();
        }
        const int total = scan[255];
        const int pos   = base + scan[tid] - pred;   // exclusive rank
        if (pred) {
            tlist[(size_t)e * TOK + pos] = t;
            wlist[(size_t)e * TOK + pos] = p0 ? w.x : w.y;
        }
        base += total;
        __syncthreads();   // protect scan[] before next chunk overwrites it
    }
    if (tid == 0) counts[e] = base;
}

// ---------------------------------------------------------------------------
// Fused expert FFN. Block = (expert, 16-token tile), 256 threads = 8 waves.
// LDS: xs[16][1024] bf16 (32 KB) + hs[16][4096] bf16 (128 KB) + tok/w (128 B).
//
// WMMA fragment gathers (A and B share contiguous k order k = 16*half + s):
//   A (16x32 bf16): lane row = lane&15; VGPR0-3 = K[k0+half*16 .. +7],
//                   VGPR4-7 = K[k0+half*16+8 .. +15]      -> 2x ds_load_b128
//   B (32x16 bf16): lane col = lane&15; lanes<16 hold K[k0..k0+15],
//                   lanes>=16 hold K[k0+16..k0+31]        -> 2x global_load_b128
//                   (weights stored transposed [N][K], K contiguous)
//   C/D (16x16 f32): VGPR r holds row r + (lane>=16)*8, col = lane&15.
// ---------------------------------------------------------------------------
__global__ __launch_bounds__(256) void moe_ffn_kernel(
    const float* __restrict__ x, const float* __restrict__ b1,
    const float* __restrict__ b2, const __bf16* __restrict__ W1T,
    const __bf16* __restrict__ W2T, const int* __restrict__ counts,
    const int* __restrict__ tlist, const float* __restrict__ wlist,
    float* __restrict__ out) {
    extern __shared__ char smem[];
    __bf16* xs    = (__bf16*)smem;                                   // [16][DDIM]
    __bf16* hs    = (__bf16*)(smem + 16 * DDIM * 2);                 // [16][HDIM]
    int*    tok_s = (int*)  (smem + 16 * DDIM * 2 + 16 * HDIM * 2);  // [16]
    float*  w_s   = (float*)(tok_s + 16);                            // [16]

    const int e     = blockIdx.y;
    const int cnt   = counts[e];
    const int tbase = blockIdx.x * 16;
    if (tbase >= cnt) return;   // uniform early-exit (before any barrier)

    const int tid = threadIdx.x;
    if (tid < 16) {
        const int r = tbase + tid;
        tok_s[tid] = (r < cnt) ? tlist[(size_t)e * TOK + r] : 0;
        w_s[tid]   = (r < cnt) ? wlist[(size_t)e * TOK + r] : 0.0f;  // padded rows add 0
    }
    __syncthreads();

    // Stage the gathered x tile into LDS as bf16 (one float4 per thread per row).
    for (int r = 0; r < 16; ++r) {
        const float4 v = ((const float4*)(x + (size_t)tok_s[r] * DDIM))[tid];
        __bf16* p = xs + r * DDIM + tid * 4;
        p[0] = (__bf16)v.x; p[1] = (__bf16)v.y; p[2] = (__bf16)v.z; p[3] = (__bf16)v.w;
    }
    __syncthreads();

    const int wave = tid >> 5;
    const int lane = tid & 31;
    const int half = lane >> 4;
    const int l15  = lane & 15;

    // -------- GEMM1: h = relu(x @ W1[e] + b1[e]),  M=16, N=HDIM, K=DDIM -----
    {
        const __bf16* W1e = W1T + (size_t)e * HDIM * DDIM;   // [HDIM][DDIM]
        const __bf16* ar  = xs + (size_t)l15 * DDIM + half * 16;
        for (int i = 0; i < 16; ++i) {                       // 2 N-tiles / iter / wave
            const int n0 = (i * 16 + wave * 2) * 16;
            const int nA = n0 + l15, nB = n0 + 16 + l15;
            const __bf16* bpA = W1e + (size_t)nA * DDIM + half * 16;
            const __bf16* bpB = W1e + (size_t)nB * DDIM + half * 16;
            v8f acc0 = {0.f,0.f,0.f,0.f,0.f,0.f,0.f,0.f};
            v8f acc1 = {0.f,0.f,0.f,0.f,0.f,0.f,0.f,0.f};
            for (int k0 = 0; k0 < DDIM; k0 += 32) {
                Frag a, fb0, fb1;
                a.u[0]   = *(const v4u*)(ar  + k0);          // ds_load_b128
                a.u[1]   = *(const v4u*)(ar  + k0 + 8);
                fb0.u[0] = *(const v4u*)(bpA + k0);          // global_load_b128
                fb0.u[1] = *(const v4u*)(bpA + k0 + 8);
                fb1.u[0] = *(const v4u*)(bpB + k0);
                fb1.u[1] = *(const v4u*)(bpB + k0 + 8);
                __builtin_prefetch(bpA + k0 + 512, 0, 1);    // global_prefetch_b8
                acc0 = wmma_bf16(a.v, fb0.v, acc0);
                acc1 = wmma_bf16(a.v, fb1.v, acc1);
            }
            const float b1A = b1[(size_t)e * HDIM + nA];
            const float b1B = b1[(size_t)e * HDIM + nB];
#pragma unroll
            for (int r = 0; r < 8; ++r) {
                const int row = r + half * 8;
                hs[(size_t)row * HDIM + nA] = (__bf16)fmaxf(acc0[r] + b1A, 0.f);
                hs[(size_t)row * HDIM + nB] = (__bf16)fmaxf(acc1[r] + b1B, 0.f);
            }
        }
    }
    __syncthreads();

    // -------- GEMM2: out += comb * (h @ W2[e] + b2[e]), N=DDIM, K=HDIM ------
    {
        const __bf16* W2e = W2T + (size_t)e * DDIM * HDIM;   // [DDIM][HDIM]
        float cw[8]; int ct[8];
#pragma unroll
        for (int r = 0; r < 8; ++r) {
            cw[r] = w_s[r + half * 8];
            ct[r] = tok_s[r + half * 8];
        }
        const __bf16* ar = hs + (size_t)l15 * HDIM + half * 16;
        for (int i = 0; i < 4; ++i) {
            const int n0 = (i * 16 + wave * 2) * 16;
            const int nA = n0 + l15, nB = n0 + 16 + l15;
            const __bf16* bpA = W2e + (size_t)nA * HDIM + half * 16;
            const __bf16* bpB = W2e + (size_t)nB * HDIM + half * 16;
            v8f acc0 = {0.f,0.f,0.f,0.f,0.f,0.f,0.f,0.f};
            v8f acc1 = {0.f,0.f,0.f,0.f,0.f,0.f,0.f,0.f};
            for (int k0 = 0; k0 < HDIM; k0 += 32) {
                Frag a, fb0, fb1;
                a.u[0]   = *(const v4u*)(ar  + k0);
                a.u[1]   = *(const v4u*)(ar  + k0 + 8);
                fb0.u[0] = *(const v4u*)(bpA + k0);
                fb0.u[1] = *(const v4u*)(bpA + k0 + 8);
                fb1.u[0] = *(const v4u*)(bpB + k0);
                fb1.u[1] = *(const v4u*)(bpB + k0 + 8);
                __builtin_prefetch(bpA + k0 + 512, 0, 1);
                acc0 = wmma_bf16(a.v, fb0.v, acc0);
                acc1 = wmma_bf16(a.v, fb1.v, acc1);
            }
            const float b2A = b2[(size_t)e * DDIM + nA];
            const float b2B = b2[(size_t)e * DDIM + nB];
#pragma unroll
            for (int r = 0; r < 8; ++r) {
                atomicAdd(out + (size_t)ct[r] * DDIM + nA, cw[r] * (acc0[r] + b2A));
                atomicAdd(out + (size_t)ct[r] * DDIM + nB, cw[r] * (acc1[r] + b2B));
            }
        }
    }
}

// ---------------------------------------------------------------------------
// Host-side launch. Inputs (setup_inputs order):
//   0:x 1:gate_W 2:gate_b 3:W1 4:b1 5:W2 6:b2 7:experts_per_token
// d_out: [ out (TOK*DDIM f32) | gate_probs (TOK*NEXP f32) ]
// ---------------------------------------------------------------------------
extern "C" void kernel_launch(void* const* d_in, const int* in_sizes, int n_in,
                              void* d_out, int out_size, void* d_ws, size_t ws_size,
                              hipStream_t stream) {
    const float* x   = (const float*)d_in[0];
    const float* gW  = (const float*)d_in[1];
    const float* gb  = (const float*)d_in[2];
    const float* W1  = (const float*)d_in[3];
    const float* b1  = (const float*)d_in[4];
    const float* W2  = (const float*)d_in[5];
    const float* b2  = (const float*)d_in[6];

    float* out        = (float*)d_out;
    float* gate_probs = out + (size_t)TOK * DDIM;

    // Workspace layout (bytes). Total ~129 MiB; bf16 weights fit the 192MB L2.
    constexpr size_t SZ_W   = (size_t)NEXP * DDIM * HDIM * sizeof(__bf16);  // 64 MiB each
    constexpr size_t OFF_W1 = 0;
    constexpr size_t OFF_W2 = OFF_W1 + SZ_W;
    constexpr size_t OFF_IX = OFF_W2 + SZ_W;                 // int2  [TOK]
    constexpr size_t OFF_WV = OFF_IX + (size_t)TOK * 8;      // float2[TOK]
    constexpr size_t OFF_CT = OFF_WV + (size_t)TOK * 8;      // int   [NEXP] (padded)
    constexpr size_t OFF_TL = OFF_CT + 256;                  // int   [NEXP][TOK]
    constexpr size_t OFF_WL = OFF_TL + (size_t)NEXP * TOK * 4; // float[NEXP][TOK]

    char*    ws    = (char*)d_ws;
    __bf16*  W1T   = (__bf16*)(ws + OFF_W1);
    __bf16*  W2T   = (__bf16*)(ws + OFF_W2);
    int2*    idx2  = (int2*)  (ws + OFF_IX);
    float2*  w2v   = (float2*)(ws + OFF_WV);
    int*     cnts  = (int*)   (ws + OFF_CT);
    int*     tlist = (int*)   (ws + OFF_TL);
    float*   wlist = (float*) (ws + OFF_WL);

    // 1) zero the accumulated output region (epilogue uses atomic adds)
    zero_out_kernel<<<(TOK * DDIM) / (256 * 4), 256, 0, stream>>>(out);

    // 2) weights f32 -> bf16, transposed to [N][K] for contiguous B-fragments
    transpose_bf16_kernel<<<dim3(HDIM / 32, DDIM / 32, NEXP), dim3(32, 8), 0, stream>>>(
        W1, W1T, DDIM, HDIM);
    transpose_bf16_kernel<<<dim3(DDIM / 32, HDIM / 32, NEXP), dim3(32, 8), 0, stream>>>(
        W2, W2T, HDIM, DDIM);

    // 3) gating + routing tables
    gate_kernel<<<TOK / 8, 256, 0, stream>>>(x, gW, gb, gate_probs, idx2, w2v);

    // 4) deterministic per-expert token buckets
    bucket_kernel<<<NEXP, 256, 0, stream>>>(idx2, w2v, cnts, tlist, wlist);

    // 5) fused expert FFN (160 KB dynamic LDS per block; CDNA5 WGP has 320 KB)
    constexpr int SMEM = 16 * DDIM * 2 + 16 * HDIM * 2 + 16 * 4 + 16 * 4;
    (void)hipFuncSetAttribute(reinterpret_cast<const void*>(moe_ffn_kernel),
                              hipFuncAttributeMaxDynamicSharedMemorySize, SMEM);
    moe_ffn_kernel<<<dim3(TOK / 16, NEXP), 256, SMEM, stream>>>(
        x, b1, b2, W1T, W2T, cnts, tlist, wlist, out);
}
